// AttentionPairBias_12103217840568
// MI455X (gfx1250) — matrix operand order repaired and checked by reading, not compiled, not run on hardware
//
#include <hip/hip_runtime.h>
#include <hip/hip_bf16.h>

// ---------------------------------------------------------------------------
// AttentionPairBias for MI455X (gfx1250, wave32, WMMA)
// Memory-bound on streaming z (512 MB, read once, non-temporal).
// All GEMMs: v_wmma_f32_16x16x32_bf16 (bf16 in, f32 accumulate).
// ---------------------------------------------------------------------------

typedef __bf16 bf16_t;
typedef __attribute__((ext_vector_type(16))) __bf16 v16bf;
typedef __attribute__((ext_vector_type(8)))  __bf16 v8bf;
typedef __attribute__((ext_vector_type(4)))  __bf16 v4bf;
typedef __attribute__((ext_vector_type(8)))  float  v8f;
typedef __attribute__((ext_vector_type(4)))  float  f4;

#define N_TOK 1024
#define CS    384
#define CZ    128
#define NH    4
#define HD    96
#define LNEPS 1e-5f
#define INV_SQRT_HD 0.10206207261596576f

// ---------------- WMMA fragment helpers (ISA 7.12.2 layouts) ----------------

// A fragment: 16x32 bf16, row-major [M][K] memory.
// lane L holds row M = L%16; element e -> K = kbase + (e/8)*16 + (L/16)*8 + e%8
__device__ __forceinline__ v16bf load_a_bf16(const bf16_t* __restrict__ base,
                                             int ld, int row0, int kbase) {
  const int lane = threadIdx.x & 31;
  const bf16_t* p = base + (size_t)(row0 + (lane & 15)) * ld + kbase + ((lane >> 4) << 3);
  v8bf lo = *(const v8bf*)p;
  v8bf hi = *(const v8bf*)(p + 16);
  v16bf f;
#pragma unroll
  for (int e = 0; e < 8; ++e) { f[e] = lo[e]; f[e + 8] = hi[e]; }
  return f;
}

// B fragment: 32x16 bf16, row-major [K][N] memory.
// lane L holds col N = nbase + L%16; element e -> K = kbase + (e/8)*16 + (L/16)*8 + e%8
__device__ __forceinline__ v16bf load_b_bf16(const bf16_t* __restrict__ base,
                                             int ld, int kbase, int nbase) {
  const int lane = threadIdx.x & 31;
  const bf16_t* p = base + (size_t)(kbase + ((lane >> 4) << 3)) * ld + nbase + (lane & 15);
  v16bf f;
#pragma unroll
  for (int e = 0; e < 8; ++e) { f[e] = p[(size_t)e * ld]; f[e + 8] = p[(size_t)(e + 16) * ld]; }
  return f;
}

__device__ __forceinline__ v8f wmma_bf16(v16bf a, v16bf b, v8f c) {
  // (neg_a, A, neg_b, B, c_mod, C, reuse_a, reuse_b)
  return __builtin_amdgcn_wmma_f32_16x16x32_bf16(false, a, false, b, (short)0, c,
                                                 false, false);
}

// C/D 16x16 f32: element r -> M = m0tile + r + (lane/16)*8, N = lane%16

// ---------------- Kernel A: layer_norm(s) -> bf16 -----------------------------
__global__ void ln_s_kernel(const float* __restrict__ s,
                            const float* __restrict__ w,
                            const float* __restrict__ b,
                            bf16_t* __restrict__ snbf) {
  const int lane = threadIdx.x & 31;
  const int wv   = threadIdx.x >> 5;
  const int row  = blockIdx.x * 4 + wv;
  const float* x = s + (size_t)row * CS;
  float v[12], s1 = 0.f, s2 = 0.f;
#pragma unroll
  for (int t = 0; t < 12; ++t) {
    v[t] = x[lane + t * 32];
    s1 += v[t];
    s2 += v[t] * v[t];
  }
#pragma unroll
  for (int m = 16; m; m >>= 1) {
    s1 += __shfl_xor(s1, m, 32);
    s2 += __shfl_xor(s2, m, 32);
  }
  const float mu = s1 * (1.f / CS);
  const float rs = rsqrtf(s2 * (1.f / CS) - mu * mu + LNEPS);
  bf16_t* o = snbf + (size_t)row * CS;
#pragma unroll
  for (int t = 0; t < 12; ++t) {
    const int k = lane + t * 32;
    o[k] = (bf16_t)((v[t] - mu) * rs * w[k] + b[k]);
  }
}

// ---------------- Kernel: transpose f32 [rows][cols] -> bf16 [cols][rows] -----
__global__ void transpose_w_kernel(const float* __restrict__ w,
                                   bf16_t* __restrict__ out,
                                   int rows, int cols) {
  const int idx = blockIdx.x * blockDim.x + threadIdx.x;
  if (idx >= rows * cols) return;
  const int n = idx / cols, k = idx - n * cols;
  out[(size_t)k * rows + n] = (bf16_t)w[idx];
}

// ---------------- Kernel B: QKV GEMM (1024x1152 = snbf @ qkv_wT) --------------
// scatter into qbuf[h][i][d] (pre-scaled), ktbuf[h][d][j], vbuf[h][j][d]
__global__ void qkv_gemm_kernel(const bf16_t* __restrict__ snbf,
                                const bf16_t* __restrict__ wT,   // [384][1152]
                                const float* __restrict__ qkv_b,
                                bf16_t* __restrict__ qbuf,
                                bf16_t* __restrict__ ktbuf,
                                bf16_t* __restrict__ vbuf) {
  const int lane = threadIdx.x & 31;
  const int wg   = blockIdx.x * 4 + (threadIdx.x >> 5); // 4608 waves
  const int tm = wg / 72, tn = wg % 72;
  v8f acc = {};
#pragma unroll
  for (int kt = 0; kt < 12; ++kt) {
    v16bf a = load_a_bf16(snbf, CS, tm * 16, kt * 32);
    v16bf b = load_b_bf16(wT, 3 * CS, kt * 32, tn * 16);
    acc = wmma_bf16(a, b, acc);
  }
  const int m0 = tm * 16 + ((lane >> 4) << 3);
  const int n  = tn * 16 + (lane & 15);
  const int sect = n / CS;           // 0=q 1=k 2=v
  const int nn = n - sect * CS;
  const int h = nn / HD, d = nn - h * HD;
  const float bias = qkv_b[n];
#pragma unroll
  for (int r = 0; r < 8; ++r) {
    const int m = m0 + r;
    const float val = acc[r] + bias;
    if (sect == 0)      qbuf [(size_t)(h * N_TOK + m) * HD + d]   = (bf16_t)(val * INV_SQRT_HD);
    else if (sect == 1) ktbuf[(size_t)(h * HD + d) * N_TOK + m]   = (bf16_t)val;
    else                vbuf [(size_t)(h * N_TOK + m) * HD + d]   = (bf16_t)val;
  }
}

// ---------------- Kernel D: stream z, LN + bias projection --------------------
// one wave per (i,j) pair; NT loads (512B coalesced per wave); writes bias
// into attnbuf[h][i][j] which the logits kernel consumes as WMMA C operand.
__global__ void zbias_kernel(const float* __restrict__ z,
                             const float* __restrict__ lnw,
                             const float* __restrict__ lnb,
                             const float* __restrict__ bw,   // [4][128]
                             const float* __restrict__ bb,   // [4]
                             float* __restrict__ attnbuf) {
  const int lane = threadIdx.x & 31;
  const size_t pair = (size_t)blockIdx.x * 8 + (threadIdx.x >> 5); // < 1M
  const float* zp = z + pair * CZ + lane * 4;
  f4 z4 = __builtin_nontemporal_load((const f4*)zp);
  float s1 = z4[0] + z4[1] + z4[2] + z4[3];
  float s2 = z4[0] * z4[0] + z4[1] * z4[1] + z4[2] * z4[2] + z4[3] * z4[3];
#pragma unroll
  for (int m = 16; m; m >>= 1) {
    s1 += __shfl_xor(s1, m, 32);
    s2 += __shfl_xor(s2, m, 32);
  }
  const float mu = s1 * (1.f / CZ);
  const float rs = rsqrtf(s2 * (1.f / CZ) - mu * mu + LNEPS);
  f4 w4 = *(const f4*)(lnw + lane * 4);
  f4 b4 = *(const f4*)(lnb + lane * 4);
  f4 ln;
#pragma unroll
  for (int c = 0; c < 4; ++c) ln[c] = (z4[c] - mu) * rs * w4[c] + b4[c];
  float hsum[NH];
#pragma unroll
  for (int h = 0; h < NH; ++h) {
    f4 bw4 = *(const f4*)(bw + h * CZ + lane * 4);
    hsum[h] = ln[0] * bw4[0] + ln[1] * bw4[1] + ln[2] * bw4[2] + ln[3] * bw4[3];
  }
#pragma unroll
  for (int m = 16; m; m >>= 1) {
#pragma unroll
    for (int h = 0; h < NH; ++h) hsum[h] += __shfl_xor(hsum[h], m, 32);
  }
  if (lane < NH) {
    const float v = (lane == 0) ? hsum[0] : (lane == 1) ? hsum[1]
                  : (lane == 2) ? hsum[2] : hsum[3];
    attnbuf[((size_t)lane << 20) + pair] = v + bb[lane];
  }
}

// ---------------- Kernel C: logits = Q @ K^T + bias (bias preloaded as C) -----
__global__ void logits_kernel(const bf16_t* __restrict__ qbuf,
                              const bf16_t* __restrict__ ktbuf,
                              float* __restrict__ attnbuf) {
  const int lane = threadIdx.x & 31;
  const int wg = blockIdx.x * 4 + (threadIdx.x >> 5); // 16384 waves
  const int head = wg >> 12;
  const int rem = wg & 4095;
  const int tm = rem >> 6, tn = rem & 63;
  const bf16_t* qh = qbuf + (size_t)head * N_TOK * HD;
  const bf16_t* kh = ktbuf + (size_t)head * HD * N_TOK;
  float* ab = attnbuf + ((size_t)head << 20);
  const int m0 = tm * 16 + ((lane >> 4) << 3);
  const int n  = tn * 16 + (lane & 15);
  v8f acc;
#pragma unroll
  for (int r = 0; r < 8; ++r) acc[r] = ab[(size_t)(m0 + r) * N_TOK + n];
#pragma unroll
  for (int kt = 0; kt < 3; ++kt) {
    v16bf a = load_a_bf16(qh, HD, tm * 16, kt * 32);
    v16bf b = load_b_bf16(kh, N_TOK, kt * 32, tn * 16);
    acc = wmma_bf16(a, b, acc);
  }
#pragma unroll
  for (int r = 0; r < 8; ++r) ab[(size_t)(m0 + r) * N_TOK + n] = acc[r];
}

// ---------------- Kernel E: softmax over j, emit bf16 probabilities -----------
__global__ void softmax_kernel(const float* __restrict__ attnbuf,
                               bf16_t* __restrict__ pbuf) {
  __shared__ float red[256];
  const int t = threadIdx.x;
  const size_t row = blockIdx.x; // [h][i]
  const float* x = attnbuf + (row << 10);
  bf16_t* p = pbuf + (row << 10);
  f4 v = *(const f4*)(x + t * 4);
  float m = fmaxf(fmaxf(v[0], v[1]), fmaxf(v[2], v[3]));
  red[t] = m;
  __syncthreads();
  for (int s = 128; s > 0; s >>= 1) {
    if (t < s) red[t] = fmaxf(red[t], red[t + s]);
    __syncthreads();
  }
  const float rowmax = red[0];
  __syncthreads();
  f4 e;
  float sum = 0.f;
#pragma unroll
  for (int c = 0; c < 4; ++c) { e[c] = __expf(v[c] - rowmax); sum += e[c]; }
  red[t] = sum;
  __syncthreads();
  for (int s = 128; s > 0; s >>= 1) {
    if (t < s) red[t] += red[t + s];
    __syncthreads();
  }
  const float inv = 1.f / red[0];
  v4bf pv;
#pragma unroll
  for (int c = 0; c < 4; ++c) pv[c] = (bf16_t)(e[c] * inv);
  *(v4bf*)(p + t * 4) = pv;
}

// ---------------- Kernel F: out = P @ V (per head), A reused over 6 N-tiles ---
__global__ void av_gemm_kernel(const bf16_t* __restrict__ pbuf,
                               const bf16_t* __restrict__ vbuf,
                               bf16_t* __restrict__ obuf) {
  const int lane = threadIdx.x & 31;
  const int wg = blockIdx.x * 4 + (threadIdx.x >> 5); // 256 waves
  const int head = wg >> 6;
  const int tm = wg & 63;
  const bf16_t* ph = pbuf + ((size_t)head << 20);
  const bf16_t* vh = vbuf + (size_t)head * N_TOK * HD;
  v8f acc[6];
#pragma unroll
  for (int tn = 0; tn < 6; ++tn) acc[tn] = {};
  for (int kt = 0; kt < 32; ++kt) {
    v16bf a = load_a_bf16(ph, N_TOK, tm * 16, kt * 32);
#pragma unroll
    for (int tn = 0; tn < 6; ++tn) {
      v16bf b = load_b_bf16(vh, HD, kt * 32, tn * 16);
      acc[tn] = wmma_bf16(a, b, acc[tn]);
    }
  }
  const int m0 = tm * 16 + ((lane >> 4) << 3);
  const int nl = lane & 15;
#pragma unroll
  for (int tn = 0; tn < 6; ++tn)
#pragma unroll
    for (int r = 0; r < 8; ++r)
      obuf[(size_t)(m0 + r) * CS + head * HD + tn * 16 + nl] = (bf16_t)acc[tn][r];
}

// ---------------- Kernel G: fused gate/out GEMMs + sigmoid + residual ---------
__global__ void final_kernel(const bf16_t* __restrict__ snbf,
                             const bf16_t* __restrict__ obuf,
                             const bf16_t* __restrict__ gate_wT, // [384][384]
                             const bf16_t* __restrict__ out_wT,  // [384][384]
                             const float* __restrict__ gate_b,
                             const float* __restrict__ out_b,
                             const float* __restrict__ s,
                             float* __restrict__ out) {
  const int lane = threadIdx.x & 31;
  const int wg = blockIdx.x * 4 + (threadIdx.x >> 5); // 1536 waves
  const int tm = wg / 24, tn = wg % 24;
  v8f accg = {}, acco = {};
#pragma unroll
  for (int kt = 0; kt < 12; ++kt) {
    v16bf ag = load_a_bf16(snbf, CS, tm * 16, kt * 32);
    v16bf bg = load_b_bf16(gate_wT, CS, kt * 32, tn * 16);
    accg = wmma_bf16(ag, bg, accg);
    v16bf ao = load_a_bf16(obuf, CS, tm * 16, kt * 32);
    v16bf bo = load_b_bf16(out_wT, CS, kt * 32, tn * 16);
    acco = wmma_bf16(ao, bo, acco);
  }
  const int m0 = tm * 16 + ((lane >> 4) << 3);
  const int n  = tn * 16 + (lane & 15);
  const float gb = gate_b[n], ob = out_b[n];
#pragma unroll
  for (int r = 0; r < 8; ++r) {
    const int m = m0 + r;
    const float g = 1.f / (1.f + __expf(-(accg[r] + gb)));
    const float o = acco[r] + ob;
    out[(size_t)m * CS + n] = s[(size_t)m * CS + n] + g * o;
  }
}

// ---------------- launch ------------------------------------------------------
extern "C" void kernel_launch(void* const* d_in, const int* in_sizes, int n_in,
                              void* d_out, int out_size, void* d_ws, size_t ws_size,
                              hipStream_t stream) {
  const float* s      = (const float*)d_in[0];
  const float* z      = (const float*)d_in[1];
  const float* ln_s_w = (const float*)d_in[2];
  const float* ln_s_b = (const float*)d_in[3];
  const float* ln_z_w = (const float*)d_in[4];
  const float* ln_z_b = (const float*)d_in[5];
  const float* qkv_w  = (const float*)d_in[6];
  const float* qkv_b  = (const float*)d_in[7];
  const float* bias_w = (const float*)d_in[8];
  const float* bias_b = (const float*)d_in[9];
  const float* out_w  = (const float*)d_in[10];
  const float* out_b  = (const float*)d_in[11];
  const float* gate_w = (const float*)d_in[12];
  const float* gate_b = (const float*)d_in[13];

  char* w = (char*)d_ws;
  size_t off = 0;
  bf16_t* snbf   = (bf16_t*)(w + off); off += (size_t)N_TOK * CS * 2;        // 768 KB
  bf16_t* qbuf   = (bf16_t*)(w + off); off += (size_t)NH * N_TOK * HD * 2;   // 768 KB
  bf16_t* ktbuf  = (bf16_t*)(w + off); off += (size_t)NH * HD * N_TOK * 2;   // 768 KB
  bf16_t* vbuf   = (bf16_t*)(w + off); off += (size_t)NH * N_TOK * HD * 2;   // 768 KB
  bf16_t* obuf   = (bf16_t*)(w + off); off += (size_t)N_TOK * CS * 2;        // 768 KB
  bf16_t* qkvwT  = (bf16_t*)(w + off); off += (size_t)CS * 3 * CS * 2;       // 864 KB
  bf16_t* gatewT = (bf16_t*)(w + off); off += (size_t)CS * CS * 2;           // 288 KB
  bf16_t* outwT  = (bf16_t*)(w + off); off += (size_t)CS * CS * 2;           // 288 KB
  float*  attnbuf= (float*) (w + off); off += (size_t)NH * N_TOK * N_TOK * 4;// 16 MB
  bf16_t* pbuf   = (bf16_t*)(w + off); off += (size_t)NH * N_TOK * N_TOK * 2;// 8 MB

  // Phase 1: LN(s), weight transposes (independent of z stream)
  ln_s_kernel<<<N_TOK / 4, 128, 0, stream>>>(s, ln_s_w, ln_s_b, snbf);
  transpose_w_kernel<<<(3 * CS * CS + 255) / 256, 256, 0, stream>>>(qkv_w, qkvwT, 3 * CS, CS);
  transpose_w_kernel<<<(CS * CS + 255) / 256, 256, 0, stream>>>(gate_w, gatewT, CS, CS);
  transpose_w_kernel<<<(CS * CS + 255) / 256, 256, 0, stream>>>(out_w, outwT, CS, CS);

  // Phase 2: QKV projection (WMMA) and the dominant z stream (bias -> attnbuf)
  qkv_gemm_kernel<<<(64 * 72) / 4, 128, 0, stream>>>(snbf, qkvwT, qkv_b, qbuf, ktbuf, vbuf);
  zbias_kernel<<<(N_TOK * N_TOK) / 8, 256, 0, stream>>>(z, ln_z_w, ln_z_b, bias_w, bias_b, attnbuf);

  // Phase 3: logits (Q@K^T with bias as WMMA C), softmax, P@V
  logits_kernel<<<(NH * 64 * 64) / 4, 128, 0, stream>>>(qbuf, ktbuf, attnbuf);
  softmax_kernel<<<NH * N_TOK, 256, 0, stream>>>(attnbuf, pbuf);
  av_gemm_kernel<<<(NH * 64) / 4, 128, 0, stream>>>(pbuf, vbuf, obuf);

  // Phase 4: fused gate/out projection + sigmoid gating + residual
  final_kernel<<<(64 * 24) / 4, 128, 0, stream>>>(snbf, obuf, gatewT, outwT,
                                                  gate_b, out_b, s, (float*)d_out);
}
